// attention_77326591197822
// MI455X (gfx1250) — matrix operand order repaired and checked by reading, not compiled, not run on hardware
//
#include <hip/hip_runtime.h>

// Problem constants (B=1)
#define S_LEN 4096
#define EMB   1024
#define HEADS 16
#define DHEAD 64

typedef __attribute__((ext_vector_type(16))) __bf16 bf16x16;
typedef __attribute__((ext_vector_type(8)))  __bf16 bf16x8;
typedef __attribute__((ext_vector_type(8)))  float  f32x8;
typedef __attribute__((ext_vector_type(4)))  float  f32x4;
typedef __attribute__((ext_vector_type(4)))  unsigned int u32x4;
typedef __attribute__((ext_vector_type(8)))  int    i32x8;
typedef __attribute__((ext_vector_type(4)))  int    i32x4;

// Build a 16-element bf16 fragment from two contiguous 16-byte chunks.
__device__ __forceinline__ bf16x16 load_frag(const __bf16* p0, const __bf16* p1) {
  bf16x8 lo = *(const bf16x8*)p0;
  bf16x8 hi = *(const bf16x8*)p1;
  bf16x16 r;
#pragma unroll
  for (int i = 0; i < 8; ++i) { r[i] = lo[i]; r[i + 8] = hi[i]; }
  return r;
}

__device__ __forceinline__ f32x8 wmma_bf16(bf16x16 a, bf16x16 b, f32x8 c) {
  // D = A*B + C, 16x16x32 bf16 -> f32
  return __builtin_amdgcn_wmma_f32_16x16x32_bf16(false, a, false, b, (short)0, c,
                                                 false, false);
}

// ---------------------------------------------------------------------------
// TDM: async DMA of a 128(rows) x 32(k) bf16 tile (row-major source, row
// stride = tensor_k elements) into LDS at lds_off. 2-D descriptor per
// CDNA5 ISA; groups 2/3 (and the extra group of
// the 6-arg builtin) zero.
// ---------------------------------------------------------------------------
__device__ __forceinline__ void tdm_load_tile_a(const __bf16* tile, unsigned lds_off,
                                                unsigned tensor_k, unsigned tensor_m) {
  unsigned long long ga = (unsigned long long)(uintptr_t)tile;
  u32x4 g0;
  g0[0] = 1u;                                   // count=1, user mode, no gather
  g0[1] = lds_off;                              // lds_addr (bytes)
  g0[2] = (unsigned)(ga & 0xffffffffu);         // global_addr[31:0]
  g0[3] = (unsigned)((ga >> 32) & 0x01ffffffu)  // global_addr[56:32]
          | (2u << 30);                         // type = 2 ("image")
  i32x8 g1;
  g1[0] = (int)(1u << 16);                      // wg_mask=0, data_size=1 (2 bytes)
  g1[1] = (int)((tensor_k & 0xffffu) << 16);    // tensor_dim0[15:0] @ bits[31:16]
  g1[2] = (int)((tensor_k >> 16) | ((tensor_m & 0xffffu) << 16));  // dim0 hi | dim1 lo
  g1[3] = (int)((tensor_m >> 16) | (32u << 16));// dim1 hi | tile_dim0 = 32
  g1[4] = (int)(128u);                          // tile_dim1 = 128, tile_dim2 = 0
  g1[5] = (int)(tensor_k);                      // tensor_dim0_stride[31:0] = K
  g1[6] = 0;                                    // stride hi | dim1_stride lo
  g1[7] = 0;
  i32x4 z4 = {0, 0, 0, 0};
  i32x8 z8 = {0, 0, 0, 0, 0, 0, 0, 0};
  __builtin_amdgcn_tensor_load_to_lds(g0, g1, z4, z4, z8, 0);
}

// ---------------------------------------------------------------------------
// Tiled GEMM  C = A(MxK) * B(KxN) + bias, via bf16 WMMA with f32 accumulate.
// Block tile 128x64, BK=32, 256 threads = 8 waves (4 in M x 2 in N),
// each wave computes 2x2 16x16 tiles.
// EPI==0: fp32 A staged manually; QKV epilogue (Q scaled / K head-major /
//         V transposed).
// EPI==1: bf16 A staged by the Tensor Data Mover (double-buffered, overlapped
//         with compute, synced with s_wait_tensorcnt); f32 output epilogue.
// ---------------------------------------------------------------------------
template <typename AT, int EPI>
__global__ __launch_bounds__(256) void gemm_bf16_wmma(
    const AT* __restrict__ A, const float* __restrict__ B,
    const float* __restrict__ bias, int M, int N, int K,
    __bf16* __restrict__ qout, __bf16* __restrict__ kout,
    __bf16* __restrict__ vtout, float* __restrict__ fout) {
  __shared__ __align__(16) __bf16 As[2][128][32];  // row-major (m,k); [1] = TDM buf
  __shared__ __align__(16) __bf16 Bs[64][32];      // transposed (n,k)

  const int tid  = threadIdx.x;
  const int lane = tid & 31;
  const int wave = tid >> 5;
  const int wm   = wave >> 1;   // 0..3 (M)
  const int wn   = wave & 1;    // 0..1 (N)
  const int l16  = lane & 15;
  const int g    = lane >> 4;   // lane half

  const int m0 = blockIdx.y * 128;
  const int n0 = blockIdx.x * 64;

  // A loader (manual path): each thread covers one half-row (16 contiguous k)
  const int ra = tid >> 1;
  const int ca = (tid & 1) * 16;
  // B loader: each thread covers one k-row, 8 contiguous n
  const int kb_ = tid >> 3;
  const int nb  = (tid & 7) * 8;

  f32x8 acc[2][2] = {};

  if constexpr (EPI == 1) {
    // Prologue: kick off DMA of the first A tile into buffer 0.
    if (wave == 0)
      tdm_load_tile_a(A + (size_t)m0 * K, (unsigned)(uintptr_t)&As[0][0][0],
                      (unsigned)K, (unsigned)M);
  }

  int ib = 0;
  for (int kk = 0; kk < K; kk += 32, ib ^= 1) {
    // ---- stage B tile (fp32 -> bf16), transposed into (n,k) ----
    const float* bptr = B + (size_t)(kk + kb_) * N + n0 + nb;
    f32x4 b0 = *(const f32x4*)bptr;
    f32x4 b1 = *(const f32x4*)(bptr + 4);
#pragma unroll
    for (int j = 0; j < 4; ++j) {
      Bs[nb + j][kb_]     = (__bf16)b0[j];
      Bs[nb + 4 + j][kb_] = (__bf16)b1[j];
    }
    if (kk + 32 < K) __builtin_prefetch(bptr + 32 * (size_t)N, 0, 1);

    if constexpr (EPI == 1) {
      // ---- TDM pipeline: issue next tile, then wait for the current one ----
      if (wave == 0) {
        if (kk + 32 < K) {
          tdm_load_tile_a(A + (size_t)m0 * K + (kk + 32),
                          (unsigned)(uintptr_t)&As[ib ^ 1][0][0],
                          (unsigned)K, (unsigned)M);
          __builtin_amdgcn_s_wait_tensorcnt(1);  // in-order: current tile landed
        } else {
          __builtin_amdgcn_s_wait_tensorcnt(0);
        }
      }
    } else {
      // ---- stage A tile manually (convert fp32 -> bf16) ----
      const AT* aptr = A + (size_t)(m0 + ra) * K + kk + ca;
#pragma unroll
      for (int i = 0; i < 2; ++i) {
        bf16x8 v;
        f32x4 u0 = *(const f32x4*)(aptr + 8 * i);
        f32x4 u1 = *(const f32x4*)(aptr + 8 * i + 4);
#pragma unroll
        for (int j = 0; j < 4; ++j) { v[j] = (__bf16)u0[j]; v[j + 4] = (__bf16)u1[j]; }
        *(bf16x8*)&As[0][ra][ca + 8 * i] = v;
      }
    }
    __syncthreads();

    const int ab = (EPI == 1) ? ib : 0;
    // ---- fragments (per ISA 16-bit A/B VGPR layouts) ----
    bf16x16 af[2], bfr[2];
#pragma unroll
    for (int mi = 0; mi < 2; ++mi) {
      const __bf16* p = &As[ab][wm * 32 + mi * 16 + l16][0];
      af[mi] = load_frag(p + 8 * g, p + 16 + 8 * g);
    }
#pragma unroll
    for (int ni = 0; ni < 2; ++ni) {
      const __bf16* p = &Bs[wn * 32 + ni * 16 + l16][0];
      bfr[ni] = load_frag(p + 8 * g, p + 16 + 8 * g);
    }
#pragma unroll
    for (int mi = 0; mi < 2; ++mi)
#pragma unroll
      for (int ni = 0; ni < 2; ++ni)
        acc[mi][ni] = wmma_bf16(af[mi], bfr[ni], acc[mi][ni]);
    __syncthreads();
  }

  // ---- epilogue ----
#pragma unroll
  for (int mi = 0; mi < 2; ++mi) {
#pragma unroll
    for (int ni = 0; ni < 2; ++ni) {
      const int Ncol = n0 + wn * 32 + ni * 16 + l16;
      const float bval = bias[Ncol];
#pragma unroll
      for (int r = 0; r < 8; ++r) {
        const int Mrow = m0 + wm * 32 + mi * 16 + r + 8 * g;
        float v = acc[mi][ni][r] + bval;
        if constexpr (EPI == 0) {
          const int head = Ncol / 192;          // col layout: [H][3*d]
          const int rr   = Ncol - head * 192;
          if (rr < 64)        // Q, fold in 1/sqrt(d)=0.125
            qout[((size_t)head * S_LEN + Mrow) * DHEAD + rr] = (__bf16)(v * 0.125f);
          else if (rr < 128)  // K head-major [h][s][d]
            kout[((size_t)head * S_LEN + Mrow) * DHEAD + (rr - 64)] = (__bf16)v;
          else                // V transposed [h][d][s]
            vtout[((size_t)head * DHEAD + (rr - 128)) * S_LEN + Mrow] = (__bf16)v;
        } else {
          fout[(size_t)Mrow * N + Ncol] = v;
        }
      }
    }
  }
}

// ---------------------------------------------------------------------------
// Flash-style causal attention. One 16-row query tile per wave; 8 waves/WG.
// Scores: Q(16x64) x K^T via 2 WMMAs per 16-key tile; online softmax with
// 16-lane butterfly reductions; probabilities routed through a per-wave LDS
// tile to convert C-layout -> A-layout; PV via 4 WMMAs against V^T.
// ---------------------------------------------------------------------------
__global__ __launch_bounds__(256) void attn_flash_wmma(
    const __bf16* __restrict__ Qb, const __bf16* __restrict__ Kb,
    const __bf16* __restrict__ Vt, __bf16* __restrict__ Ab) {
  __shared__ __align__(16) __bf16 P[8][16][32];  // per-wave prob tile (m, key)

  const int tid  = threadIdx.x;
  const int lane = tid & 31;
  const int wave = tid >> 5;
  const int l16  = lane & 15;
  const int g    = lane >> 4;
  const int h    = blockIdx.y;
  const int q0   = blockIdx.x * 128 + wave * 16;

  const __bf16* Qh = Qb + (size_t)h * S_LEN * DHEAD;
  const __bf16* Kh = Kb + (size_t)h * S_LEN * DHEAD;
  const __bf16* Vh = Vt + (size_t)h * DHEAD * S_LEN;

  // Q fragments (reused across all key blocks); Q already holds 1/sqrt(d).
  bf16x16 qf[2];
#pragma unroll
  for (int t = 0; t < 2; ++t) {
    const __bf16* p = Qh + (size_t)(q0 + l16) * DHEAD + 32 * t;
    qf[t] = load_frag(p + 16 * g, p + 16 * g + 8);
  }

  float mrow[8], lrow[8];
  f32x8 o[4] = {};
#pragma unroll
  for (int r = 0; r < 8; ++r) { mrow[r] = -3.0e38f; lrow[r] = 0.0f; }

  for (int kb = 0; kb < q0 + 16; kb += 32) {  // causal: keys <= q0+15 needed
    f32x8 sc[2] = {};
#pragma unroll
    for (int n = 0; n < 2; ++n) {
      const int key = kb + 16 * n + l16;
      const __bf16* kp = Kh + (size_t)key * DHEAD + 16 * g;
#pragma unroll
      for (int t = 0; t < 2; ++t)
        sc[n] = wmma_bf16(qf[t], load_frag(kp + 32 * t, kp + 32 * t + 8), sc[n]);
    }
    // online softmax per row r (row = q0 + r + 8*g, col = kb + 16n + l16)
#pragma unroll
    for (int r = 0; r < 8; ++r) {
      const int row = q0 + r + 8 * g;
      float s0 = (kb + l16 <= row)      ? sc[0][r] : -1.0e6f;  // NEG_BIG
      float s1 = (kb + 16 + l16 <= row) ? sc[1][r] : -1.0e6f;
      float mx = fmaxf(s0, s1);
#pragma unroll
      for (int d = 1; d < 16; d <<= 1) mx = fmaxf(mx, __shfl_xor(mx, d, 32));
      const float mnew    = fmaxf(mrow[r], mx);
      const float rescale = __expf(mrow[r] - mnew);
      const float p0 = __expf(s0 - mnew);
      const float p1 = __expf(s1 - mnew);
      float rs = p0 + p1;
#pragma unroll
      for (int d = 1; d < 16; d <<= 1) rs += __shfl_xor(rs, d, 32);
      lrow[r] = lrow[r] * rescale + rs;
      mrow[r] = mnew;
#pragma unroll
      for (int ni = 0; ni < 4; ++ni) o[ni][r] *= rescale;
      P[wave][r + 8 * g][l16]      = (__bf16)p0;  // C-layout scatter
      P[wave][r + 8 * g][16 + l16] = (__bf16)p1;
    }
    asm volatile("s_wait_dscnt 0" ::: "memory");  // LDS store->load ordering
    const __bf16* pp = &P[wave][l16][0];
    bf16x16 pf = load_frag(pp + 16 * g, pp + 16 * g + 8);  // A-layout gather
#pragma unroll
    for (int ni = 0; ni < 4; ++ni) {
      const __bf16* vp = Vh + (size_t)(16 * ni + l16) * S_LEN + kb + 16 * g;
      o[ni] = wmma_bf16(pf, load_frag(vp, vp + 8), o[ni]);
    }
  }

  // normalize and write attention output as [s][h*64+d] (bf16)
#pragma unroll
  for (int r = 0; r < 8; ++r) {
    const int row = q0 + r + 8 * g;
    const float inv = 1.0f / lrow[r];
#pragma unroll
    for (int ni = 0; ni < 4; ++ni)
      Ab[(size_t)row * EMB + h * DHEAD + 16 * ni + l16] = (__bf16)(o[ni][r] * inv);
  }
}

// ---------------------------------------------------------------------------
extern "C" void kernel_launch(void* const* d_in, const int* in_sizes, int n_in,
                              void* d_out, int out_size, void* d_ws, size_t ws_size,
                              hipStream_t stream) {
  const float* x  = (const float*)d_in[0];  // [1,4096,1024]
  const float* W1 = (const float*)d_in[1];  // [1024,3072]
  const float* b1 = (const float*)d_in[2];  // [3072]
  const float* Wp = (const float*)d_in[3];  // [1024,1024]
  const float* bp = (const float*)d_in[4];  // [1024]
  float* out = (float*)d_out;               // [1,4096,1024] f32

  unsigned char* ws = (unsigned char*)d_ws;
  const size_t seg = (size_t)HEADS * S_LEN * DHEAD * sizeof(__bf16);  // 8 MB
  __bf16* Qb = (__bf16*)(ws);            // [H][S][d], pre-scaled by 1/sqrt(d)
  __bf16* Kb = (__bf16*)(ws + seg);      // [H][S][d]
  __bf16* Vt = (__bf16*)(ws + 2 * seg);  // [H][d][S]
  __bf16* Ab = (__bf16*)(ws + 3 * seg);  // [S][E]

  dim3 blk(256);

  // 1) QKV projection: x @ W1 + b1 -> split Q/K/V (bf16)
  gemm_bf16_wmma<float, 0><<<dim3(3 * EMB / 64, S_LEN / 128), blk, 0, stream>>>(
      x, W1, b1, S_LEN, 3 * EMB, EMB, Qb, Kb, Vt, nullptr);

  // 2) causal flash attention -> Ab (bf16)
  attn_flash_wmma<<<dim3(S_LEN / 128, HEADS), blk, 0, stream>>>(Qb, Kb, Vt, Ab);

  // 3) output projection: Ab @ Wp + bp -> out (f32), A tiles via TDM
  gemm_bf16_wmma<__bf16, 1><<<dim3(EMB / 64, S_LEN / 128), blk, 0, stream>>>(
      Ab, Wp, bp, S_LEN, EMB, EMB, nullptr, nullptr, nullptr, out);
}
